// Transformer_Block_82781199663412
// MI455X (gfx1250) — matrix operand order repaired and checked by reading, not compiled, not run on hardware
//
#include <hip/hip_runtime.h>
#include <cstdint>

#define DMODEL 2048
#define NHEAD  16
#define KVHEAD 4
#define HEADD  128
#define QKVD   3072   // D + 2*DKV
#define FFND   8192

typedef __bf16 bf16;
typedef __attribute__((ext_vector_type(8)))  __bf16 bf16x8;
typedef __attribute__((ext_vector_type(16))) __bf16 v16bf;
typedef __attribute__((ext_vector_type(8)))  float  v8f;

static __device__ __forceinline__ v16bf frag_cat(bf16x8 lo, bf16x8 hi) {
  return __builtin_shufflevector(lo, hi, 0,1,2,3,4,5,6,7,8,9,10,11,12,13,14,15);
}

// A-matrix fragment (16x32 bf16, MxK); src row-major with row stride ld.
// Lane l: M = l%16; half h=l/16; elems 0..7 -> K = h*8+0..7, elems 8..15 -> K = 16+h*8+0..7
static __device__ __forceinline__ v16bf load_frag_a(const bf16* base, int ld, int lane) {
  const int row = lane & 15, half = lane >> 4;
  const bf16* p = base + (long)row * ld + half * 8;
  bf16x8 lo = *(const bf16x8*)(p);
  bf16x8 hi = *(const bf16x8*)(p + 16);
  return frag_cat(lo, hi);
}

// B-matrix fragment (32x16 bf16, KxN); memory is row-major [n][k] with row stride ld.
// Lane l: N = l%16; half h=l/16; elems 0..7 -> K = h*8+0..7, elems 8..15 -> K = 16+h*8+0..7
static __device__ __forceinline__ v16bf load_frag_b(const bf16* base, int ld, int lane) {
  const int n = lane & 15, half = lane >> 4;
  const bf16* p = base + (long)n * ld + half * 8;
  bf16x8 lo = *(const bf16x8*)(p);
  bf16x8 hi = *(const bf16x8*)(p + 16);
  return frag_cat(lo, hi);
}

static __device__ __forceinline__ v8f wmma_bf16(v16bf a, v16bf b, v8f c) {
  return __builtin_amdgcn_wmma_f32_16x16x32_bf16(false, a, false, b, (short)0, c, false, false);
}

// Async global -> LDS copy, 16B per lane, tracked by ASYNCcnt (CDNA5).
// lds = wave-relative LDS byte offset (generic-pointer low 32 bits), g = per-lane global addr.
static __device__ __forceinline__ void async_b128(unsigned lds, const void* g) {
  asm volatile("global_load_async_to_lds_b128 %0, %1, off" :: "v"(lds), "v"(g) : "memory");
}

// ---------------------------------------------------------------------------
// GEMM: C[M,N] (f32) = A[M,K](bf16) * W[N,K](bf16)^T.
// Block = 8 waves (2M x 4N), wave tile 32x64, block tile 64x256.
// Double-buffered LDS staging via GLOBAL_LOAD_ASYNC_TO_LDS_B128:
// per K-step the block moves A 4KB + B 16KB (5 async b128 per thread) and
// executes 64 WMMAs (52 flop / global byte); tiles are reused 4x (A) / 2x (B).
// ---------------------------------------------------------------------------
__global__ __launch_bounds__(256) void k_gemm_bf16(const bf16* __restrict__ A,
                                                   const bf16* __restrict__ W,
                                                   float* __restrict__ C,
                                                   int M, int N, int K) {
  __shared__ bf16 ldsA[2][64 * 32];     // 2 x 4KB
  __shared__ bf16 ldsB[2][256 * 32];    // 2 x 16KB
  const int tid  = threadIdx.x;
  const int lane = tid & 31;
  const int wave = tid >> 5;
  const int wm = wave >> 2;             // 0..1
  const int wn = wave & 3;              // 0..3
  const int mBlock = blockIdx.y * 64;
  const int nBlock = blockIdx.x * 256;

  const unsigned ldsA0 = (unsigned)(uintptr_t)&ldsA[0][0];
  const unsigned ldsB0 = (unsigned)(uintptr_t)&ldsB[0][0];
  const int rowA = tid >> 2;            // 0..63
  const int seg  = (tid & 3) * 8;       // 0,8,16,24 (bf16 elems; 16B each)

  v8f acc[2][4];
#pragma unroll
  for (int i = 0; i < 2; ++i)
#pragma unroll
    for (int j = 0; j < 4; ++j)
#pragma unroll
      for (int r = 0; r < 8; ++r) acc[i][j][r] = 0.0f;

  auto stage = [&](int buf, int k) {
    async_b128(ldsA0 + (unsigned)(buf * (64 * 32 * 2) + (rowA * 32 + seg) * 2),
               A + (long)(mBlock + rowA) * K + k + seg);
#pragma unroll
    for (int i = 0; i < 4; ++i) {
      const int r = rowA + 64 * i;      // 0..255
      async_b128(ldsB0 + (unsigned)(buf * (256 * 32 * 2) + (r * 32 + seg) * 2),
                 W + (long)(nBlock + r) * K + k + seg);
    }
  };

  const int nsteps = K >> 5;
  stage(0, 0);
  for (int s = 0; s < nsteps; ++s) {
    const int cur = s & 1;
    if (s + 1 < nsteps) {
      stage(cur ^ 1, (s + 1) << 5);
      asm volatile("s_wait_asynccnt 5" ::: "memory");   // current buffer's 5 done (in-order)
    } else {
      asm volatile("s_wait_asynccnt 0" ::: "memory");
    }
    __syncthreads();

    v16bf a[2], b[4];
#pragma unroll
    for (int i = 0; i < 2; ++i)
      a[i] = load_frag_a(&ldsA[cur][(wm * 32 + i * 16) * 32], 32, lane);
#pragma unroll
    for (int j = 0; j < 4; ++j)
      b[j] = load_frag_b(&ldsB[cur][(wn * 64 + j * 16) * 32], 32, lane);
#pragma unroll
    for (int i = 0; i < 2; ++i)
#pragma unroll
      for (int j = 0; j < 4; ++j)
        acc[i][j] = wmma_bf16(a[i], b[j], acc[i][j]);
    __syncthreads();                    // all waves done reading `cur` before it is re-staged
  }

  const int half = lane >> 4, nl = lane & 15;
  const int mBase = mBlock + wm * 32;
  const int nBase = nBlock + wn * 64;
#pragma unroll
  for (int i = 0; i < 2; ++i)
#pragma unroll
    for (int j = 0; j < 4; ++j)
#pragma unroll
      for (int r = 0; r < 8; ++r)
        C[(long)(mBase + i * 16 + r + 8 * half) * N + (nBase + j * 16 + nl)] = acc[i][j][r];
}

// ---------------------------------------------------------------------------
// Flash attention, causal GQA. One wave handles 16 query rows of one head.
// Q [B,NH,T,HD] bf16 (l2-normed), K [B,KVH,T,HD] bf16 (l2-normed),
// Vt [B,KVH,HD,T] bf16 (transposed).  Output Y [B*T, D] bf16 (head-major).
// ---------------------------------------------------------------------------
__global__ __launch_bounds__(32) void k_attn(const bf16* __restrict__ Qn,
                                             const bf16* __restrict__ Kn,
                                             const bf16* __restrict__ Vt,
                                             bf16* __restrict__ Y, int T) {
  __shared__ bf16 Plds[16 * 32];
  const int lane = threadIdx.x;
  const int half = lane >> 4, nl = lane & 15;
  const int bh = blockIdx.x;               // b*NHEAD + nh
  const int b  = bh / NHEAD;
  const int nh = bh % NHEAD;
  const int kvh = nh % KVHEAD;             // reference: kv head = nh % KVH
  const int qbase = blockIdx.y * 16;

  const bf16* Qb = Qn + ((long)(b * NHEAD + nh) * T + qbase) * HEADD;
  const bf16* Kb = Kn + ((long)(b * KVHEAD + kvh) * T) * HEADD;
  const bf16* Vb = Vt + ((long)(b * KVHEAD + kvh) * HEADD) * T;

  v16bf qf[4];
#pragma unroll
  for (int c = 0; c < 4; ++c) qf[c] = load_frag_a(Qb + c * 32, HEADD, lane);

  v8f o[8];
  float mi[8], li[8];
#pragma unroll
  for (int j = 0; j < 8; ++j)
#pragma unroll
    for (int r = 0; r < 8; ++r) o[j][r] = 0.0f;
#pragma unroll
  for (int r = 0; r < 8; ++r) { mi[r] = -3.0e38f; li[r] = 0.0f; }

  const float scale = 0.0883883476483184f;  // 1/sqrt(HD)

  for (int kb = 0; kb < qbase + 16; kb += 32) {
    __builtin_prefetch(Kb + (long)(kb + 32) * HEADD, 0, 1);
    __builtin_prefetch(Vb + kb + 32, 0, 1);
    const bool do2 = (kb + 16) <= (qbase + 15);   // second key subtile has any valid col
    v8f s0, s1;
#pragma unroll
    for (int r = 0; r < 8; ++r) { s0[r] = 0.0f; s1[r] = 0.0f; }
#pragma unroll
    for (int c = 0; c < 4; ++c) {
      v16bf kf = load_frag_b(Kb + (long)kb * HEADD + c * 32, HEADD, lane);
      s0 = wmma_bf16(qf[c], kf, s0);
    }
    if (do2) {
#pragma unroll
      for (int c = 0; c < 4; ++c) {
        v16bf kf = load_frag_b(Kb + (long)(kb + 16) * HEADD + c * 32, HEADD, lane);
        s1 = wmma_bf16(qf[c], kf, s1);
      }
    }
    float alpha[8];
#pragma unroll
    for (int r = 0; r < 8; ++r) {
      const int gq = qbase + r + 8 * half;
      float v0 = s0[r] * scale;
      if (kb + nl > gq) v0 = -3.0e38f;          // causal mask
      float v1 = -3.0e38f;
      if (do2) { v1 = s1[r] * scale; if (kb + 16 + nl > gq) v1 = -3.0e38f; }
      float tm = fmaxf(v0, v1);
#pragma unroll
      for (int m = 1; m < 16; m <<= 1) tm = fmaxf(tm, __shfl_xor(tm, m, 32));
      const float nm = fmaxf(mi[r], tm);
      const float a  = __expf(mi[r] - nm);
      const float e0 = __expf(v0 - nm);
      const float e1 = do2 ? __expf(v1 - nm) : 0.0f;
      float rs = e0 + e1;
#pragma unroll
      for (int m = 1; m < 16; m <<= 1) rs += __shfl_xor(rs, m, 32);
      li[r] = li[r] * a + rs;
      mi[r] = nm;
      alpha[r] = a;
      Plds[(r + 8 * half) * 32 + nl]      = (bf16)e0;   // stage P row-major in LDS
      Plds[(r + 8 * half) * 32 + 16 + nl] = (bf16)e1;
    }
#pragma unroll
    for (int j = 0; j < 8; ++j)
#pragma unroll
      for (int r = 0; r < 8; ++r) o[j][r] *= alpha[r];

    v16bf pf = load_frag_a(Plds, 32, lane);             // LDS ops in-order within wave
#pragma unroll
    for (int j = 0; j < 8; ++j) {
      v16bf vf = load_frag_b(Vb + (long)(j * 16) * T + kb, T, lane);
      o[j] = wmma_bf16(pf, vf, o[j]);
    }
  }

#pragma unroll
  for (int j = 0; j < 8; ++j)
#pragma unroll
    for (int r = 0; r < 8; ++r) {
      const float ov = o[j][r] / li[r];
      Y[(long)(b * T + qbase + r + 8 * half) * DMODEL + nh * HEADD + j * 16 + nl] = (bf16)ov;
    }
}

// ---------------------------------------------------------------------------
// qkv postprocess: *conv_w, l2norm q/k heads, layout for attention (V transposed)
// grid = (B*T, 24), block = 32 (one wave per 128-elem head chunk)
// ---------------------------------------------------------------------------
__global__ __launch_bounds__(32) void k_qkv_post(const float* __restrict__ qkv,
                                                 const float* __restrict__ conv_w,
                                                 bf16* __restrict__ Qn,
                                                 bf16* __restrict__ Kn,
                                                 bf16* __restrict__ Vt, int T) {
  const int row = blockIdx.x;            // b*T + t
  const int c   = blockIdx.y;            // 0..23: 16 q heads, 4 k heads, 4 v heads
  const int lane = threadIdx.x;
  const int b = row / T, t = row % T;
  const float* src = qkv + (long)row * QKVD + c * HEADD + lane * 4;
  const float* cw  = conv_w + c * HEADD + lane * 4;
  float v[4]; float ss = 0.0f;
#pragma unroll
  for (int e = 0; e < 4; ++e) { v[e] = src[e] * cw[e]; ss += v[e] * v[e]; }
#pragma unroll
  for (int m = 1; m < 32; m <<= 1) ss += __shfl_xor(ss, m, 32);
  if (c < 20) {                           // l2norm for q and k heads
    const float inv = 1.0f / fmaxf(sqrtf(ss), 1e-5f);
#pragma unroll
    for (int e = 0; e < 4; ++e) v[e] *= inv;
  }
  if (c < 16) {
    bf16* dst = Qn + ((long)(b * NHEAD + c) * T + t) * HEADD + lane * 4;
#pragma unroll
    for (int e = 0; e < 4; ++e) dst[e] = (bf16)v[e];
  } else if (c < 20) {
    bf16* dst = Kn + ((long)(b * KVHEAD + (c - 16)) * T + t) * HEADD + lane * 4;
#pragma unroll
    for (int e = 0; e < 4; ++e) dst[e] = (bf16)v[e];
  } else {
    bf16* dst = Vt + ((long)(b * KVHEAD + (c - 20)) * HEADD) * T + t;
#pragma unroll
    for (int e = 0; e < 4; ++e) dst[(long)(lane * 4 + e) * T] = (bf16)v[e];
  }
}

// ---------------------------------------------------------------------------
static __device__ __forceinline__ float block_sum256(float v) {
  __shared__ float red[8];
  const int lane = threadIdx.x & 31, w = threadIdx.x >> 5;
#pragma unroll
  for (int m = 1; m < 32; m <<= 1) v += __shfl_xor(v, m, 32);
  if (lane == 0) red[w] = v;
  __syncthreads();
  float t = 0.0f;
#pragma unroll
  for (int i = 0; i < 8; ++i) t += red[i];
  return t;
}

__global__ __launch_bounds__(256) void k_rms_bf16(const float* __restrict__ X,
                                                  const float* __restrict__ w,
                                                  bf16* __restrict__ outp) {
  const long row = blockIdx.x;
  const float* p = X + row * DMODEL;
  float vals[8]; float ss = 0.0f;
#pragma unroll
  for (int j = 0; j < 8; ++j) { int i = threadIdx.x + j * 256; vals[j] = p[i]; ss += vals[j] * vals[j]; }
  const float tot = block_sum256(ss);
  const float inv = rsqrtf(tot * (1.0f / DMODEL) + 1e-5f);
#pragma unroll
  for (int j = 0; j < 8; ++j) { int i = threadIdx.x + j * 256; outp[row * DMODEL + i] = (bf16)(vals[j] * w[i] * inv); }
}

__global__ __launch_bounds__(256) void k_addres_rms(const float* __restrict__ qin,
                                                    const float* __restrict__ yp,
                                                    const float* __restrict__ w,
                                                    float* __restrict__ q2,
                                                    bf16* __restrict__ h) {
  const long row = blockIdx.x;
  float vals[8]; float ss = 0.0f;
#pragma unroll
  for (int j = 0; j < 8; ++j) {
    int i = threadIdx.x + j * 256;
    float v = qin[row * DMODEL + i] + yp[row * DMODEL + i];
    vals[j] = v; ss += v * v;
  }
  const float tot = block_sum256(ss);
  const float inv = rsqrtf(tot * (1.0f / DMODEL) + 1e-5f);
#pragma unroll
  for (int j = 0; j < 8; ++j) {
    int i = threadIdx.x + j * 256;
    q2[row * DMODEL + i] = vals[j];
    h[row * DMODEL + i]  = (bf16)(vals[j] * w[i] * inv);
  }
}

__global__ __launch_bounds__(256) void k_cvt_bf16(const float* __restrict__ in,
                                                  bf16* __restrict__ outp, long n) {
  long i = (long)blockIdx.x * 256 + threadIdx.x;
  const long stride = (long)gridDim.x * 256;
  for (; i < n; i += stride) outp[i] = (bf16)in[i];
}

__global__ __launch_bounds__(256) void k_gelu_bf16(const float* __restrict__ in,
                                                   bf16* __restrict__ outp, long n) {
  long i = (long)blockIdx.x * 256 + threadIdx.x;
  const long stride = (long)gridDim.x * 256;
  for (; i < n; i += stride) {
    float x = in[i];
    outp[i] = (bf16)(0.5f * x * (1.0f + erff(x * 0.7071067811865475f)));  // exact gelu
  }
}

__global__ __launch_bounds__(256) void k_final_add(float* __restrict__ o,
                                                   const float* __restrict__ a, long n) {
  long i = (long)blockIdx.x * 256 + threadIdx.x;
  const long stride = (long)gridDim.x * 256;
  for (; i < n; i += stride) o[i] += a[i];
}

// ---------------------------------------------------------------------------
extern "C" void kernel_launch(void* const* d_in, const int* in_sizes, int n_in,
                              void* d_out, int out_size, void* d_ws, size_t ws_size,
                              hipStream_t stream) {
  (void)n_in; (void)out_size; (void)ws_size;
  const float* q      = (const float*)d_in[0];
  const float* rms1_w = (const float*)d_in[1];
  const float* w_qkv  = (const float*)d_in[2];
  const float* conv_w = (const float*)d_in[3];
  const float* w_ap   = (const float*)d_in[4];
  const float* rms2_w = (const float*)d_in[5];
  const float* w_fc   = (const float*)d_in[6];
  const float* w_fp   = (const float*)d_in[7];
  float* out = (float*)d_out;

  const long MT = (long)in_sizes[0] / DMODEL;  // B*T = 4096
  const int  T  = (int)(MT / 2);               // B = 2

  // Workspace layout (256B aligned). `big` is temporally reused for all four
  // f32 GEMM outputs (qkv -> yproj -> act -> ffn); xh_bf is reused x -> h.
  char* ws = (char*)d_ws; size_t off = 0;
  auto alloc = [&](size_t bytes) -> char* {
    char* p = ws + off; off = (off + bytes + 255) & ~(size_t)255; return p;
  };
  bf16*  wqkv_bf = (bf16*)alloc((size_t)QKVD  * DMODEL * 2);
  bf16*  wap_bf  = (bf16*)alloc((size_t)DMODEL * DMODEL * 2);
  bf16*  wfc_bf  = (bf16*)alloc((size_t)FFND  * DMODEL * 2);
  bf16*  wfp_bf  = (bf16*)alloc((size_t)DMODEL * FFND  * 2);
  float* big     = (float*)alloc((size_t)MT * FFND * 4);
  bf16*  xh_bf   = (bf16*)alloc((size_t)MT * DMODEL * 2);
  bf16*  y_bf    = (bf16*)alloc((size_t)MT * DMODEL * 2);
  bf16*  qn_bf   = (bf16*)alloc((size_t)MT * DMODEL * 2);
  bf16*  kn_bf   = (bf16*)alloc((size_t)MT * 512 * 2);
  bf16*  vt_bf   = (bf16*)alloc((size_t)MT * 512 * 2);
  bf16*  gact_bf = (bf16*)alloc((size_t)MT * FFND * 2);

  auto cvt = [&](const float* s, bf16* d, long n) {
    long blocks = (n + 255) / 256; if (blocks > 8192) blocks = 8192;
    k_cvt_bf16<<<dim3((unsigned)blocks), dim3(256), 0, stream>>>(s, d, n);
  };
  cvt(w_qkv, wqkv_bf, (long)QKVD * DMODEL);
  cvt(w_ap,  wap_bf,  (long)DMODEL * DMODEL);
  cvt(w_fc,  wfc_bf,  (long)FFND * DMODEL);
  cvt(w_fp,  wfp_bf,  (long)DMODEL * FFND);

  // x = rmsnorm(q) -> bf16
  k_rms_bf16<<<dim3((unsigned)MT), dim3(256), 0, stream>>>(q, rms1_w, xh_bf);
  // qkv = x @ w_qkv^T
  k_gemm_bf16<<<dim3(QKVD / 256, (unsigned)(MT / 64)), dim3(256), 0, stream>>>(
      xh_bf, wqkv_bf, big, (int)MT, QKVD, DMODEL);
  // conv_w, l2norm, layout
  k_qkv_post<<<dim3((unsigned)MT, 24), dim3(32), 0, stream>>>(big, conv_w, qn_bf, kn_bf, vt_bf, T);
  // flash attention
  k_attn<<<dim3(2 * NHEAD, T / 16), dim3(32), 0, stream>>>(qn_bf, kn_bf, vt_bf, y_bf, T);
  // y @ w_attn_proj^T
  k_gemm_bf16<<<dim3(DMODEL / 256, (unsigned)(MT / 64)), dim3(256), 0, stream>>>(
      y_bf, wap_bf, big, (int)MT, DMODEL, DMODEL);
  // q2 = q + y; h = rmsnorm(q2); q2 -> d_out
  k_addres_rms<<<dim3((unsigned)MT), dim3(256), 0, stream>>>(q, big, rms2_w, out, xh_bf);
  // h @ w_fc^T
  k_gemm_bf16<<<dim3(FFND / 256, (unsigned)(MT / 64)), dim3(256), 0, stream>>>(
      xh_bf, wfc_bf, big, (int)MT, FFND, DMODEL);
  // exact gelu -> bf16
  k_gelu_bf16<<<dim3(8192), dim3(256), 0, stream>>>(big, gact_bf, MT * (long)FFND);
  // gelu(h) @ w_ffn_proj^T
  k_gemm_bf16<<<dim3(DMODEL / 256, (unsigned)(MT / 64)), dim3(256), 0, stream>>>(
      gact_bf, wfp_bf, big, (int)MT, DMODEL, FFND);
  // out = q2 + ffn
  k_final_add<<<dim3(4096), dim3(256), 0, stream>>>(out, big, MT * (long)DMODEL);
}
